// MoEFSCIL_19688130085040
// MI455X (gfx1250) — matrix operand order repaired and checked by reading, not compiled, not run on hardware
//
#include <hip/hip_runtime.h>
#include <hip/hip_bf16.h>

typedef __attribute__((ext_vector_type(16))) _Float16 v16h;
typedef __attribute__((ext_vector_type(8)))  _Float16 v8h;
typedef __attribute__((ext_vector_type(8)))  float    v8f;

#define NB    64      // batch
#define NL    49      // H*W
#define NDIM  512     // dim == d_inner
#define NE    4       // experts
#define NK    4       // scan directions
#define NN    64      // d_state
#define NR    64      // dt_rank
#define NP    3136    // NB*NL positions
#define N2D   1024    // 2*D
#define NRDBL 192     // R + 2N

// ---------------- helpers ----------------

__device__ __forceinline__ int dir_map(int k, int t) {
    // sequence index t of direction k -> spatial l = h*7+w
    int t2 = (k & 2) ? (48 - t) : t;
    return (k & 1) ? ((t2 % 7) * 7 + t2 / 7) : t2;
}

// A-fragment (16x32 f16, row-major source, K contiguous):
// lane m = lane&15, half h = lane>>4 -> 16 contiguous K at 16h
__device__ __forceinline__ v16h load_afrag(const _Float16* row, int k0, int hh) {
    v8h lo = *(const v8h*)(row + k0 + hh * 16);
    v8h hi = *(const v8h*)(row + k0 + hh * 16 + 8);
    v16h a;
#pragma unroll
    for (int i = 0; i < 8; ++i) { a[i] = lo[i]; a[i + 8] = hi[i]; }
    return a;
}

// B-fragment (32x16 f16) from B^T rows (N-major, K contiguous):
// lane n = lane&15, half h = lane>>4 -> 16 contiguous K at 16h
__device__ __forceinline__ v16h load_bfrag(const _Float16* btrow, int k0, int hh) {
    return *(const v16h*)(btrow + k0 + hh * 16);
}

// ---------------- K0: global average pool for gating ----------------
__global__ void pool_x_kernel(const float* __restrict__ x, float* __restrict__ x_flat) {
    int b = blockIdx.x, d = threadIdx.x;
    float s = 0.f;
    for (int l = 0; l < NL; ++l) s += x[(size_t)(b * NL + l) * NDIM + d];
    x_flat[b * NDIM + d] = s * (1.f / 49.f);
}

// ---------------- K1: gate (softmax, top-2, capacity, aux) ----------------
__global__ void gate_kernel(const float* __restrict__ x_flat,
                            const float* __restrict__ wg, const float* __restrict__ bg,
                            float* __restrict__ top_scores, int* __restrict__ top_idx,
                            float* __restrict__ auxv) {
    __shared__ float raw[NB][NE];
    __shared__ float msk[NB][NE];
    __shared__ float colsum[NE], rawmean[NE], maskmean[NE];
    int tid = threadIdx.x;            // 256 threads: (b,e) pairs
    int b = tid >> 2, e = tid & 3;
    float acc = bg[e];
    for (int c = 0; c < NDIM; ++c) acc += x_flat[b * NDIM + c] * wg[e * NDIM + c];
    raw[b][e] = acc;
    __syncthreads();
    if (tid < NB) {
        float m = raw[tid][0];
        for (int j = 1; j < NE; ++j) m = fmaxf(m, raw[tid][j]);
        float s = 0.f, v[NE];
        for (int j = 0; j < NE; ++j) { v[j] = __expf(raw[tid][j] - m); s += v[j]; }
        for (int j = 0; j < NE; ++j) { raw[tid][j] = v[j] / s; msk[tid][j] = 0.f; }
        int i1 = 0;
        for (int j = 1; j < NE; ++j) if (raw[tid][j] > raw[tid][i1]) i1 = j;
        int i2 = -1;
        for (int j = 0; j < NE; ++j) { if (j == i1) continue; if (i2 < 0 || raw[tid][j] > raw[tid][i2]) i2 = j; }
        msk[tid][i1] = 1.f; msk[tid][i2] = 1.f;
    }
    __syncthreads();
    if (tid < NE) {
        float cs = 0.f, rm = 0.f, mm = 0.f;
        for (int bb = 0; bb < NB; ++bb) {
            cs += raw[bb][tid] * msk[bb][tid];
            rm += raw[bb][tid];
            mm += msk[bb][tid];
        }
        colsum[tid] = cs; rawmean[tid] = rm / (float)NB; maskmean[tid] = mm / (float)NB;
    }
    __syncthreads();
    if (tid == 0) {
        float a = 0.f;
        for (int j = 0; j < NE; ++j) { float d = maskmean[j] - rawmean[j]; a += d * d; }
        *auxv = 0.01f * (a / (float)NE);
    }
    if (tid < NB) {
        const float capacity = 80.f;  // int(1.25*64)
        float gs[NE];
        for (int j = 0; j < NE; ++j)
            gs[j] = raw[tid][j] * msk[tid][j] / (colsum[j] + 1e-6f) * capacity;
        int i1 = 0;
        for (int j = 1; j < NE; ++j) if (gs[j] > gs[i1]) i1 = j;
        int i2 = (i1 == 0) ? 1 : 0;
        for (int j = 0; j < NE; ++j) { if (j == i1) continue; if (gs[j] > gs[i2]) i2 = j; }
        top_scores[tid * 2 + 0] = gs[i1]; top_scores[tid * 2 + 1] = gs[i2];
        top_idx[tid * 2 + 0] = i1;        top_idx[tid * 2 + 1] = i2;
    }
}

// ---------------- f32 -> f16 conversion ----------------
__global__ void cvt_f16_kernel(const float* __restrict__ s, _Float16* __restrict__ d, int n) {
    int i = blockIdx.x * blockDim.x + threadIdx.x;
    if (i < n) d[i] = (_Float16)s[i];
}

// ---------------- K2: in_proj GEMM  xz[e][p][1024] = x[p][512] @ W[e]^T ----------------
__global__ void gemm_inproj_kernel(const _Float16* __restrict__ xA,
                                   const _Float16* __restrict__ wB,
                                   float* __restrict__ xz) {
    int e = blockIdx.z, mt = blockIdx.y, nt = blockIdx.x;   // m: position tile, n: out-chan tile
    int lane = threadIdx.x;
    int n = lane & 15, hh = lane >> 4;
    const _Float16* Arow = xA + (size_t)(mt * 16 + n) * NDIM;
    const _Float16* Brow = wB + (size_t)e * N2D * NDIM + (size_t)(nt * 16 + n) * NDIM;
    v8f c = {};
    for (int k0 = 0; k0 < NDIM; k0 += 32) {
        __builtin_prefetch(Brow + k0 + 64, 0, 1);
        v16h a = load_afrag(Arow, k0, hh);
        v16h b = load_bfrag(Brow, k0, hh);
        c = __builtin_amdgcn_wmma_f32_16x16x32_f16(false, a, false, b, (short)0, c, false, false);
    }
    float* o = xz + (size_t)e * NP * N2D;
#pragma unroll
    for (int v = 0; v < 8; ++v)
        o[(size_t)(mt * 16 + hh * 8 + v) * N2D + nt * 16 + n] = c[v];
}

// ---------------- K3: depthwise 3x3 conv + bias + SiLU ----------------
__global__ void conv_silu_kernel(const float* __restrict__ xz,
                                 const float* __restrict__ cw, const float* __restrict__ cb,
                                 float* __restrict__ xcf32, _Float16* __restrict__ xcf16) {
    int l = blockIdx.x, b = blockIdx.y, e = blockIdx.z, d = threadIdx.x;
    int h = l / 7, w = l % 7;
    float acc = cb[e * NDIM + d];
#pragma unroll
    for (int kh = 0; kh < 3; ++kh) {
        int hh = h + kh - 1;
        if (hh < 0 || hh >= 7) continue;
#pragma unroll
        for (int kw = 0; kw < 3; ++kw) {
            int ww = w + kw - 1;
            if (ww < 0 || ww >= 7) continue;
            acc += xz[((size_t)e * NP + b * NL + hh * 7 + ww) * N2D + d] *
                   cw[((e * NDIM + d) * 3 + kh) * 3 + kw];
        }
    }
    float s = acc / (1.f + __expf(-acc));       // SiLU
    size_t idx = ((size_t)e * NP + b * NL + l) * NDIM + d;
    xcf32[idx] = s;
    xcf16[idx] = (_Float16)s;
}

// ---------------- K4: x_proj GEMM per (e,k): x_dbl[p][192] = W[192,512] @ xs ----------------
__global__ void gemm_xproj_kernel(const _Float16* __restrict__ wA,
                                  const _Float16* __restrict__ xcB,
                                  float* __restrict__ xdbl, _Float16* __restrict__ xdbl16) {
    int ek = blockIdx.z, rt = blockIdx.y, pt = blockIdx.x;
    int e = ek >> 2, k = ek & 3;
    int lane = threadIdx.x;
    int n = lane & 15, hh = lane >> 4;
    const _Float16* Arow = wA + ((size_t)ek * NRDBL + rt * 16 + n) * NDIM;
    int p = pt * 16 + n;
    int b = p / NL, t = p % NL;
    int l = dir_map(k, t);                            // direction gather folded into B address
    const _Float16* Brow = xcB + ((size_t)e * NP + b * NL + l) * NDIM;
    v8f c = {};
    for (int k0 = 0; k0 < NDIM; k0 += 32) {
        v16h a = load_afrag(Arow, k0, hh);
        v16h bf = load_bfrag(Brow, k0, hh);
        c = __builtin_amdgcn_wmma_f32_16x16x32_f16(false, a, false, bf, (short)0, c, false, false);
    }
    size_t pbase = (size_t)ek * NP + pt * 16 + n;
#pragma unroll
    for (int v = 0; v < 8; ++v) {
        int r = rt * 16 + hh * 8 + v;
        xdbl[pbase * NRDBL + r] = c[v];
        if (r < NR) xdbl16[pbase * NR + r] = (_Float16)c[v];   // dt rows, f16 for next GEMM
    }
}

// ---------------- K5: dt_proj GEMM + bias + softplus -> dt[e,k,d,p] (f16) ----------------
__global__ void gemm_dtproj_kernel(const _Float16* __restrict__ wA,
                                   const _Float16* __restrict__ xB,
                                   const float* __restrict__ dtb,
                                   _Float16* __restrict__ dtout) {
    int ek = blockIdx.z, dt_ = blockIdx.y, pt = blockIdx.x;
    int lane = threadIdx.x;
    int n = lane & 15, hh = lane >> 4;
    const _Float16* Arow = wA + ((size_t)ek * NDIM + dt_ * 16 + n) * NR;
    const _Float16* Brow = xB + ((size_t)ek * NP + pt * 16 + n) * NR;
    v8f c = {};
    for (int k0 = 0; k0 < NR; k0 += 32) {
        v16h a = load_afrag(Arow, k0, hh);
        v16h b = load_bfrag(Brow, k0, hh);
        c = __builtin_amdgcn_wmma_f32_16x16x32_f16(false, a, false, b, (short)0, c, false, false);
    }
#pragma unroll
    for (int v = 0; v < 8; ++v) {
        int d = dt_ * 16 + hh * 8 + v;
        float val = c[v] + dtb[ek * NDIM + d];
        float sp = (val > 20.f) ? val : log1pf(__expf(val));   // softplus
        dtout[((size_t)ek * NDIM + d) * NP + pt * 16 + n] = (_Float16)sp;
    }
}

// ---------------- K6: selective scan, one block per (e,b,k), thread = d ----------------
__global__ void __launch_bounds__(512, 1)
scan_kernel(const _Float16* __restrict__ dt, const float* __restrict__ xconv,
            const float* __restrict__ xdbl, const float* __restrict__ A_log,
            const float* __restrict__ Ds, float* __restrict__ ydir) {
    int blk = blockIdx.x;                // e*256 + b*4 + k
    int k = blk & 3, b = (blk >> 2) & 63, e = blk >> 8;
    int d = threadIdx.x;
    int ek = e * 4 + k;
    __shared__ float sB[NN], sC[NN];
    float A[NN], hst[NN];
#pragma unroll
    for (int nn = 0; nn < NN; ++nn) {
        A[nn] = -__expf(A_log[((size_t)ek * NDIM + d) * NN + nn]);
        hst[nn] = 0.f;
    }
    float Dskip = Ds[ek * NDIM + d];
    const float* xcb = xconv + (size_t)e * NP * NDIM;
    const _Float16* dtrow = dt + ((size_t)ek * NDIM + d) * NP;
    for (int t = 0; t < NL; ++t) {
        int p = b * NL + t;
        int l = dir_map(k, t);
        if (d < 2 * NN) {
            float v = xdbl[((size_t)ek * NP + p) * NRDBL + NR + d];
            if (d < NN) sB[d] = v; else sC[d - NN] = v;
        }
        __syncthreads();
        float dtv = (float)dtrow[p];
        float u   = xcb[(size_t)(b * NL + l) * NDIM + d];
        float dtu = dtv * u;
        float y = 0.f;
#pragma unroll
        for (int nn = 0; nn < NN; ++nn) {
            float dA = __expf(dtv * A[nn]);
            hst[nn] = dA * hst[nn] + dtu * sB[nn];
            y += hst[nn] * sC[nn];
        }
        ydir[((size_t)ek * NDIM + d) * NP + p] = y + u * Dskip;
        __syncthreads();
    }
}

// ---------------- K7: combine 4 directions + out_norm LN + SiLU(z) gate ----------------
__global__ void combine_kernel(const float* __restrict__ ydir, const float* __restrict__ xz,
                               const float* __restrict__ ong, const float* __restrict__ onb,
                               float* __restrict__ ygated) {
    int l = blockIdx.x, b = blockIdx.y, e = blockIdx.z, d = threadIdx.x;
    int lt = (l % 7) * 7 + l / 7;
    size_t row = (size_t)e * 4;
    float y = ydir[((row + 0) * NDIM + d) * NP + b * NL + l]
            + ydir[((row + 2) * NDIM + d) * NP + b * NL + (48 - l)]
            + ydir[((row + 1) * NDIM + d) * NP + b * NL + lt]
            + ydir[((row + 3) * NDIM + d) * NP + b * NL + (48 - lt)];
    __shared__ float red[NDIM];
    red[d] = y; __syncthreads();
    for (int s = NDIM / 2; s > 0; s >>= 1) { if (d < s) red[d] += red[d + s]; __syncthreads(); }
    float mean = red[0] * (1.f / NDIM);
    __syncthreads();
    float dv = y - mean;
    red[d] = dv * dv; __syncthreads();
    for (int s = NDIM / 2; s > 0; s >>= 1) { if (d < s) red[d] += red[d + s]; __syncthreads(); }
    float var = red[0] * (1.f / NDIM);
    float normed = dv * rsqrtf(var + 1e-5f) * ong[e * NDIM + d] + onb[e * NDIM + d];
    float z = xz[((size_t)e * NP + b * NL + l) * N2D + NDIM + d];
    float sil = z / (1.f + __expf(-z));
    ygated[(((size_t)e * NB + b) * NL + l) * NDIM + d] = normed * sil;
}

// ---------------- K8: pool over L + expert LN ----------------
__global__ void pool_ln_kernel(const float* __restrict__ ygated,
                               const float* __restrict__ ng, const float* __restrict__ nb,
                               float* __restrict__ eouts) {
    int b = blockIdx.x, e = blockIdx.y, d = threadIdx.x;
    float s = 0.f;
    for (int l = 0; l < NL; ++l) s += ygated[(((size_t)e * NB + b) * NL + l) * NDIM + d];
    float pooled = s * (1.f / 49.f);
    __shared__ float red[NDIM];
    red[d] = pooled; __syncthreads();
    for (int st = NDIM / 2; st > 0; st >>= 1) { if (d < st) red[d] += red[d + st]; __syncthreads(); }
    float mean = red[0] * (1.f / NDIM);
    __syncthreads();
    float dv = pooled - mean;
    red[d] = dv * dv; __syncthreads();
    for (int st = NDIM / 2; st > 0; st >>= 1) { if (d < st) red[d] += red[d + st]; __syncthreads(); }
    float var = red[0] * (1.f / NDIM);
    eouts[((size_t)e * NB + b) * NDIM + d] =
        dv * rsqrtf(var + 1e-5f) * ng[e * NDIM + d] + nb[e * NDIM + d];
}

// ---------------- K9: top-2 mixing + aux ----------------
__global__ void mix_kernel(const float* __restrict__ eouts,
                           const float* __restrict__ tscores, const int* __restrict__ tidx,
                           const float* __restrict__ auxv, float* __restrict__ out) {
    int b = blockIdx.x, d = threadIdx.x;
    float v = 0.f;
#pragma unroll
    for (int j = 0; j < 2; ++j) {
        int ei = tidx[b * 2 + j];
        v += tscores[b * 2 + j] * eouts[((size_t)ei * NB + b) * NDIM + d];
    }
    out[b * NDIM + d] = v;
    if (b == 0 && d == 0) out[NB * NDIM] = *auxv;   // aux scalar
}

// ---------------- host ----------------
extern "C" void kernel_launch(void* const* d_in, const int* in_sizes, int n_in,
                              void* d_out, int out_size, void* d_ws, size_t ws_size,
                              hipStream_t stream) {
    (void)in_sizes; (void)n_in; (void)out_size; (void)ws_size;
    const float* x         = (const float*)d_in[0];
    const float* w_gate_w  = (const float*)d_in[1];
    const float* w_gate_b  = (const float*)d_in[2];
    const float* in_proj_w = (const float*)d_in[3];
    const float* conv_w    = (const float*)d_in[4];
    const float* conv_b    = (const float*)d_in[5];
    const float* x_proj_w  = (const float*)d_in[6];
    const float* dt_proj_w = (const float*)d_in[7];
    const float* dt_proj_b = (const float*)d_in[8];
    const float* A_log     = (const float*)d_in[9];
    const float* Ds        = (const float*)d_in[10];
    const float* out_norm_g = (const float*)d_in[11];
    const float* out_norm_b = (const float*)d_in[12];
    const float* norm_g    = (const float*)d_in[13];
    const float* norm_b    = (const float*)d_in[14];
    float* out = (float*)d_out;

    char* base = (char*)d_ws;
    size_t off = 0;
    auto take = [&](size_t bytes) -> char* {
        char* p = base + off;
        off = (off + bytes + 255) & ~(size_t)255;
        return p;
    };
    float*     x_flat  = (float*)take((size_t)NB * NDIM * 4);
    float*     tscores = (float*)take((size_t)NB * 2 * 4);
    int*       tidx    = (int*)take((size_t)NB * 2 * 4);
    float*     auxv    = (float*)take(4);
    _Float16*  x16     = (_Float16*)take((size_t)NP * NDIM * 2);
    _Float16*  win16   = (_Float16*)take((size_t)NE * N2D * NDIM * 2);
    _Float16*  wxp16   = (_Float16*)take((size_t)NE * NK * NRDBL * NDIM * 2);
    _Float16*  wdt16   = (_Float16*)take((size_t)NE * NK * NDIM * NR * 2);
    float*     xz      = (float*)take((size_t)NE * NP * N2D * 4);
    float*     xconv   = (float*)take((size_t)NE * NP * NDIM * 4);
    _Float16*  xconv16 = (_Float16*)take((size_t)NE * NP * NDIM * 2);
    float*     xdbl    = (float*)take((size_t)NE * NK * NP * NRDBL * 4);
    _Float16*  xdbl16  = (_Float16*)take((size_t)NE * NK * NP * NR * 2);
    _Float16*  dtbuf   = (_Float16*)take((size_t)NE * NK * NDIM * NP * 2);
    float*     ydir    = (float*)take((size_t)NE * NK * NDIM * NP * 4);
    float*     ygated  = (float*)take((size_t)NE * NP * NDIM * 4);
    float*     eouts   = (float*)take((size_t)NE * NB * NDIM * 4);

    pool_x_kernel<<<dim3(NB), dim3(NDIM), 0, stream>>>(x, x_flat);
    gate_kernel<<<dim3(1), dim3(256), 0, stream>>>(x_flat, w_gate_w, w_gate_b,
                                                   tscores, tidx, auxv);
    {
        int n1 = NP * NDIM;
        cvt_f16_kernel<<<(n1 + 255) / 256, 256, 0, stream>>>(x, x16, n1);
        int n2 = NE * N2D * NDIM;
        cvt_f16_kernel<<<(n2 + 255) / 256, 256, 0, stream>>>(in_proj_w, win16, n2);
        int n3 = NE * NK * NRDBL * NDIM;
        cvt_f16_kernel<<<(n3 + 255) / 256, 256, 0, stream>>>(x_proj_w, wxp16, n3);
        int n4 = NE * NK * NDIM * NR;
        cvt_f16_kernel<<<(n4 + 255) / 256, 256, 0, stream>>>(dt_proj_w, wdt16, n4);
    }
    gemm_inproj_kernel<<<dim3(N2D / 16, NP / 16, NE), dim3(32), 0, stream>>>(x16, win16, xz);
    conv_silu_kernel<<<dim3(NL, NB, NE), dim3(NDIM), 0, stream>>>(xz, conv_w, conv_b,
                                                                  xconv, xconv16);
    gemm_xproj_kernel<<<dim3(NP / 16, NRDBL / 16, NE * NK), dim3(32), 0, stream>>>(
        wxp16, xconv16, xdbl, xdbl16);
    gemm_dtproj_kernel<<<dim3(NP / 16, NDIM / 16, NE * NK), dim3(32), 0, stream>>>(
        wdt16, xdbl16, dt_proj_b, dtbuf);
    scan_kernel<<<dim3(NE * NB * NK), dim3(NDIM), 0, stream>>>(dtbuf, xconv, xdbl,
                                                               A_log, Ds, ydir);
    combine_kernel<<<dim3(NL, NB, NE), dim3(NDIM), 0, stream>>>(ydir, xz, out_norm_g,
                                                                out_norm_b, ygated);
    pool_ln_kernel<<<dim3(NB, NE), dim3(NDIM), 0, stream>>>(ygated, norm_g, norm_b, eouts);
    mix_kernel<<<dim3(NB), dim3(NDIM), 0, stream>>>(eouts, tscores, tidx, auxv, out);
}
